// VPSAwithAtt_11149735100592
// MI455X (gfx1250) — matrix operand order repaired and checked by reading, not compiled, not run on hardware
//
#include <hip/hip_runtime.h>
#include <hip/hip_bf16.h>

// ---------------------------------------------------------------------------
// Types for CDNA5 WMMA (wave32, v_wmma_f32_16x16x32_bf16)
// ---------------------------------------------------------------------------
typedef __attribute__((ext_vector_type(16))) __bf16         v16bf;
typedef __attribute__((ext_vector_type(8)))  float          v8f;
typedef __attribute__((ext_vector_type(8)))  unsigned short ushort8;
typedef unsigned short u16;

#define N_PTS   4096
#define L_CHL   512
#define IMG_CHL 1024
#define HEADS   4
#define HW      256
#define KEY_M   (HEADS * L_CHL)   // 2048

static __device__ __forceinline__ u16 f32_to_bf16(float f) {
    unsigned int u = __float_as_uint(f);
    // round-to-nearest-even truncation to bf16
    unsigned int r = (u + 0x7FFFu + ((u >> 16) & 1u)) >> 16;
    return (u16)r;
}
static __device__ __forceinline__ float bf16_to_f32(u16 h) {
    return __uint_as_float(((unsigned int)h) << 16);
}

// ---------------------------------------------------------------------------
// 1) Bilinear interpolation from BEV grid -> imgB bf16 [N_PTS][IMG_CHL]
//    (point-major so it can be the B operand of the keys GEMM)
//    threads: id -> n = id>>10, c = id&1023 ; lanes share keypoint n.
// ---------------------------------------------------------------------------
__global__ __launch_bounds__(256)
void interp_kernel(const float* __restrict__ kp, const float* __restrict__ bev,
                   u16* __restrict__ imgB) {
    unsigned int id = blockIdx.x * 256u + threadIdx.x;   // 4096*1024 threads
    int n = id >> 10;
    int c = id & (IMG_CHL - 1);

    float x = kp[n * 4 + 1] * 256.0f;   // /VOXEL /BEV_STRIDE, PC_MIN=0
    float y = kp[n * 4 + 2] * 256.0f;
    int xf = (int)floorf(x);
    int yf = (int)floorf(y);
    int x0 = min(max(xf, 0), HW - 1);
    int x1 = min(max(xf + 1, 0), HW - 1);
    int y0 = min(max(yf, 0), HW - 1);
    int y1 = min(max(yf + 1, 0), HW - 1);

    const float* ch = bev + (size_t)c * (HW * HW);
    float Ia = ch[y0 * HW + x0];
    float Ib = ch[y1 * HW + x0];
    float Ic = ch[y0 * HW + x1];
    float Id = ch[y1 * HW + x1];

    float x0f = (float)x0, x1f = (float)x1, y0f = (float)y0, y1f = (float)y1;
    float wa = (x1f - x) * (y1f - y);
    float wb = (x1f - x) * (y - y0f);
    float wc = (x - x0f) * (y1f - y);
    float wd = (x - x0f) * (y - y0f);

    imgB[(size_t)n * IMG_CHL + c] = f32_to_bf16(Ia * wa + Ib * wb + Ic * wc + Id * wd);
}

// ---------------------------------------------------------------------------
// 2) point_features (f32 [512][4096]) -> pfB bf16 [4096][512]
// ---------------------------------------------------------------------------
__global__ __launch_bounds__(256)
void transpose_pf_kernel(const float* __restrict__ pf, u16* __restrict__ pfB) {
    unsigned int id = blockIdx.x * 256u + threadIdx.x;   // 4096*512 threads
    int n = id >> 9;
    int c = id & (L_CHL - 1);
    pfB[(size_t)n * L_CHL + c] = f32_to_bf16(pf[(size_t)c * N_PTS + n]);
}

// generic f32 -> bf16 convert (weights)
__global__ __launch_bounds__(256)
void cvt_kernel(const float* __restrict__ in, u16* __restrict__ out, int count) {
    int i = blockIdx.x * 256 + threadIdx.x;
    if (i < count) out[i] = f32_to_bf16(in[i]);
}

// ---------------------------------------------------------------------------
// 3) WMMA GEMM:  D[M][N] = A[M][K] * B[N][K]^T + bias[M]
//    A row-major bf16, B point-major bf16 (each lane's fragment contiguous).
//    One wave computes a 16x128 tile (8 accumulators); 4 waves / block cover
//    64 rows. Per-lane fragment layout used here (contiguous k, not the ISA table order):
//      A 16x32: m = lane&15; K = 16*(lane>>4) + e
//      B 32x16: n = lane&15; K = 16*(lane>>4) + e
//      C/D    : n = lane&15; M = r + 8*(lane>>4) for VGPR r
// ---------------------------------------------------------------------------
#define GEMM_NT 8   // 8 n-tiles of 16 => 128 columns per wave

__global__ __launch_bounds__(128)
void wmma_gemm_kernel(const u16* __restrict__ A, const u16* __restrict__ B,
                      const float* __restrict__ bias,
                      float* __restrict__ Df32, u16* __restrict__ Dbf16,
                      int M, int N, int K) {
    const int lane = threadIdx.x & 31;
    const int wave = threadIdx.x >> 5;
    const int half = lane >> 4;
    const int lm   = lane & 15;

    const int tm  = (blockIdx.x * 4 + wave) * 16;   // row tile base
    const int tn0 = blockIdx.y * (16 * GEMM_NT);    // col strip base
    if (tm >= M) return;

    v8f acc[GEMM_NT] = {};

    const size_t arow = (size_t)(tm + lm) * K;
    for (int kk = 0; kk < K; kk += 32) {
        union { v16bf v; ushort8 h[2]; } a;
        a.h[0] = *(const ushort8*)(A + arow + kk + 16 * half);
        a.h[1] = *(const ushort8*)(A + arow + kk + 16 * half + 8);
#pragma unroll
        for (int t = 0; t < GEMM_NT; ++t) {
            const int n = tn0 + t * 16 + lm;
            v16bf b = *(const v16bf*)(B + (size_t)n * K + kk + 16 * half);
            acc[t] = __builtin_amdgcn_wmma_f32_16x16x32_bf16(
                false, a.v, false, b, (short)0, acc[t], false, false);
        }
    }

#pragma unroll
    for (int t = 0; t < GEMM_NT; ++t) {
        const int n = tn0 + t * 16 + lm;
#pragma unroll
        for (int r = 0; r < 8; ++r) {
            const int m = tm + r + 8 * half;
            float v = acc[t][r] + (bias ? bias[m] : 0.0f);
            if (Df32)  Df32[(size_t)m * N + n] = v;
            if (Dbf16) Dbf16[(size_t)m * N + n] = f32_to_bf16(v);
        }
    }
}

// ---------------------------------------------------------------------------
// 4) score + softmax over heads -> weightmap (written straight into d_out)
//    one thread per point n; keys reads coalesced over n.
// ---------------------------------------------------------------------------
__global__ __launch_bounds__(256)
void score_softmax_kernel(const u16* __restrict__ keys, const float* __restrict__ Lq,
                          float* __restrict__ out_wm) {
    int n = blockIdx.x * 256 + threadIdx.x;   // 0..4095
    float s[HEADS] = {0.f, 0.f, 0.f, 0.f};
    for (int l = 0; l < L_CHL; ++l) {
        float q = Lq[(size_t)l * N_PTS + n];
#pragma unroll
        for (int h = 0; h < HEADS; ++h)
            s[h] += bf16_to_f32(keys[((size_t)(h * L_CHL + l)) * N_PTS + n]) * q;
    }
    const float inv_sqrt_dk = 0.044194173824159216f;   // 1/sqrt(512)
    float mx = -3.0e38f;
#pragma unroll
    for (int h = 0; h < HEADS; ++h) { s[h] *= inv_sqrt_dk; mx = fmaxf(mx, s[h]); }
    float sum = 0.f;
#pragma unroll
    for (int h = 0; h < HEADS; ++h) { s[h] = __expf(s[h] - mx); sum += s[h]; }
    float rs = 1.0f / sum;
#pragma unroll
    for (int h = 0; h < HEADS; ++h) out_wm[(size_t)h * N_PTS + n] = s[h] * rs;
}

// ---------------------------------------------------------------------------
// 5) t = L_query + sum_h wm[h]*keys[h]  +  per-channel mean / variance
//    one block per channel c (512 blocks), reduce over n.
// ---------------------------------------------------------------------------
__global__ __launch_bounds__(256)
void fuse_t_kernel(const u16* __restrict__ keys, const float* __restrict__ Lq,
                   const float* __restrict__ wm, float* __restrict__ tbuf,
                   float* __restrict__ mu, float* __restrict__ var) {
    __shared__ float ssum[256];
    __shared__ float ssq[256];
    const int c = blockIdx.x;
    float sum = 0.f, sq = 0.f;
    for (int n = threadIdx.x; n < N_PTS; n += 256) {
        float v = Lq[(size_t)c * N_PTS + n];
#pragma unroll
        for (int h = 0; h < HEADS; ++h)
            v += wm[(size_t)h * N_PTS + n] *
                 bf16_to_f32(keys[((size_t)(h * L_CHL + c)) * N_PTS + n]);
        tbuf[(size_t)c * N_PTS + n] = v;
        sum += v;
        sq  += v * v;
    }
    ssum[threadIdx.x] = sum;
    ssq[threadIdx.x]  = sq;
    __syncthreads();
    for (int s2 = 128; s2 > 0; s2 >>= 1) {
        if ((int)threadIdx.x < s2) {
            ssum[threadIdx.x] += ssum[threadIdx.x + s2];
            ssq[threadIdx.x]  += ssq[threadIdx.x + s2];
        }
        __syncthreads();
    }
    if (threadIdx.x == 0) {
        float m = ssum[0] * (1.0f / N_PTS);
        mu[c]  = m;
        var[c] = ssq[0] * (1.0f / N_PTS) - m * m;
    }
}

// ---------------------------------------------------------------------------
// 6) LayerNorm apply + transpose to bf16 [N][K] for the fusion GEMM
// ---------------------------------------------------------------------------
__global__ __launch_bounds__(256)
void normalize_t_kernel(const float* __restrict__ tbuf, const float* __restrict__ mu,
                        const float* __restrict__ var, const float* __restrict__ lnw,
                        const float* __restrict__ lnb, u16* __restrict__ tnB) {
    unsigned int id = blockIdx.x * 256u + threadIdx.x;   // 4096*512 threads
    int n = id >> 9;
    int c = id & (L_CHL - 1);
    float v = (tbuf[(size_t)c * N_PTS + n] - mu[c]) * rsqrtf(var[c] + 1e-5f);
    v = v * lnw[n] + lnb[n];
    tnB[(size_t)n * L_CHL + c] = f32_to_bf16(v);
}

// ---------------------------------------------------------------------------
// launch
// ---------------------------------------------------------------------------
extern "C" void kernel_launch(void* const* d_in, const int* in_sizes, int n_in,
                              void* d_out, int out_size, void* d_ws, size_t ws_size,
                              hipStream_t stream) {
    const float* pf   = (const float*)d_in[0];   // (512, 4096)
    const float* kp   = (const float*)d_in[1];   // (4096, 4)
    const float* bev  = (const float*)d_in[2];   // (1024, 256, 256)
    const float* wL   = (const float*)d_in[3];   // (512, 512)
    const float* bL   = (const float*)d_in[4];   // (512,)
    const float* wImg = (const float*)d_in[5];   // (4, 512, 1024)
    const float* bImg = (const float*)d_in[6];   // (4, 512)
    const float* lnw  = (const float*)d_in[7];   // (4096,)
    const float* lnb  = (const float*)d_in[8];   // (4096,)
    const float* wF   = (const float*)d_in[9];   // (512, 512)
    const float* bF   = (const float*)d_in[10];  // (512,)
    float* out = (float*)d_out;                  // [512*4096 fusion | 4*4096 wm]

    // ---- workspace layout (all sizes 256B-aligned) ----
    char* ws = (char*)d_ws;
    size_t off = 0;
    u16* imgB  = (u16*)(ws + off); off += (size_t)N_PTS * IMG_CHL * 2;     // 8 MB
    u16* pfB   = (u16*)(ws + off); off += (size_t)N_PTS * L_CHL * 2;       // 4 MB
    u16* wImgB = (u16*)(ws + off); off += (size_t)KEY_M * IMG_CHL * 2;     // 4 MB
    u16* wLB   = (u16*)(ws + off); off += (size_t)L_CHL * L_CHL * 2;       // 0.5 MB
    u16* wFB   = (u16*)(ws + off); off += (size_t)L_CHL * L_CHL * 2;       // 0.5 MB
    u16* keysB = (u16*)(ws + off); off += (size_t)KEY_M * N_PTS * 2;       // 16 MB
    float* Lq  = (float*)(ws + off); off += (size_t)L_CHL * N_PTS * 4;     // 8 MB
    float* tb  = (float*)(ws + off); off += (size_t)L_CHL * N_PTS * 4;     // 8 MB
    u16* tnB   = (u16*)(ws + off); off += (size_t)N_PTS * L_CHL * 2;       // 4 MB
    float* mu  = (float*)(ws + off); off += L_CHL * 4;
    float* var = (float*)(ws + off); off += L_CHL * 4;
    (void)ws_size; (void)in_sizes; (void)n_in; (void)out_size;

    float* out_wm = out + (size_t)L_CHL * N_PTS;

    // 1) bilinear gather -> bf16 [N][IMG_CHL]
    interp_kernel<<<(N_PTS * IMG_CHL) / 256, 256, 0, stream>>>(kp, bev, imgB);
    // 2) transposes / weight conversions
    transpose_pf_kernel<<<(N_PTS * L_CHL) / 256, 256, 0, stream>>>(pf, pfB);
    cvt_kernel<<<(KEY_M * IMG_CHL + 255) / 256, 256, 0, stream>>>(wImg, wImgB, KEY_M * IMG_CHL);
    cvt_kernel<<<(L_CHL * L_CHL + 255) / 256, 256, 0, stream>>>(wL, wLB, L_CHL * L_CHL);
    cvt_kernel<<<(L_CHL * L_CHL + 255) / 256, 256, 0, stream>>>(wF, wFB, L_CHL * L_CHL);
    // 3) keys = img_head_w x imgB  (2048 x 4096, K=1024) -> bf16
    wmma_gemm_kernel<<<dim3(KEY_M / 64, N_PTS / 128), 128, 0, stream>>>(
        wImgB, imgB, bImg, nullptr, keysB, KEY_M, N_PTS, IMG_CHL);
    //    L_query = conv_L_w x pfB  (512 x 4096, K=512) -> f32
    wmma_gemm_kernel<<<dim3(L_CHL / 64, N_PTS / 128), 128, 0, stream>>>(
        wLB, pfB, bL, Lq, nullptr, L_CHL, N_PTS, L_CHL);
    // 4) attention scores + softmax over heads -> weightmap in d_out
    score_softmax_kernel<<<N_PTS / 256, 256, 0, stream>>>(keysB, Lq, out_wm);
    // 5) t = z + L_query, per-channel mean/var
    fuse_t_kernel<<<L_CHL, 256, 0, stream>>>(keysB, Lq, out_wm, tb, mu, var);
    // 6) LayerNorm + transpose to [N][K] bf16
    normalize_t_kernel<<<(N_PTS * L_CHL) / 256, 256, 0, stream>>>(tb, mu, var, lnw, lnb, tnB);
    // 7) fusion = conv_fusion_w x tnorm -> d_out (f32, [512][4096])
    wmma_gemm_kernel<<<dim3(L_CHL / 64, N_PTS / 128), 128, 0, stream>>>(
        wFB, tnB, bF, out, nullptr, L_CHL, N_PTS, L_CHL);
}
